// JTFST1D_46523085750862
// MI455X (gfx1250) — matrix operand order repaired and checked by reading, not compiled, not run on hardware
//
#include <hip/hip_runtime.h>
#include <math.h>
#include <stdint.h>

// ---------------------------------------------------------------------------
// JTFST on MI455X (gfx1250, wave32, WMMA + async-LDS + TDM).
// Long 1-D convolutions are Toeplitz GEMMs on v_wmma_f32_16x16x32_f16.
// Signal windows are staged via global_load_async_to_lds_b128; the second-
// order wavelet window is DMA'd by the Tensor Data Mover (tensor_load_to_lds).
// ---------------------------------------------------------------------------

typedef _Float16 h16;
typedef __attribute__((ext_vector_type(16))) _Float16 v16h;
typedef __attribute__((ext_vector_type(8)))  float    v8f;
typedef __attribute__((ext_vector_type(4)))  float    f32x4;
typedef __attribute__((ext_vector_type(4), aligned(4))) float f32x4u; // dword-aligned 16B load
typedef __attribute__((ext_vector_type(4)))  int      i32x4;
typedef __attribute__((ext_vector_type(8)))  int      i32x8;

#define SRATE   48000.0f
#define T_LEN   16384
#define NW1     192
#define PI_M025 0.7511255444649425f   // pi^(-1/4)
#define TWOPI   6.2831853071795864769f

#define XPAD    24576                 // zero padding (halves) each side of x
#define XH_STRIDE (T_LEN + 2 * XPAD)  // 65536 halves per batch row
#define SPAD    12352                 // zero padding each side of scalogram rows
#define TS      (T_LEN + 2 * SPAD)    // 41088 halves per scalogram row

struct Cfg1 { int Kb[12]; int Mb[12]; unsigned off[12]; };
struct Cfg2 { int K2[12]; int M2[12]; int K2s[12]; unsigned off[12]; int nb[12]; };
struct WF   { float re[2][25]; float im[2][25]; };

// ---------------------------------------------------------------------------
// x (f32) -> zero-padded f16 signal rows (async-copy source for scat1).
// ---------------------------------------------------------------------------
__global__ void xcast(const float* __restrict__ x, h16* __restrict__ xh) {
  int id = blockIdx.x * 256 + threadIdx.x;          // 2 * 65536 ids
  int row = id >> 16, p = id & 65535;
  int idx = p - XPAD;
  float v = (idx >= 0 && idx < T_LEN) ? x[row * T_LEN + idx] : 0.0f;
  xh[(size_t)row * XH_STRIDE + p] = (h16)v;
}

// Zero the padded scalogram buffer (pads must read as zero in scat2).
__global__ void zscal(h16* __restrict__ scal, int n8) {
  int id = blockIdx.x * 256 + threadIdx.x;
  if (id < n8) { f32x4 z = {}; *(f32x4*)(scal + (size_t)id * 8) = z; }
}

// ---------------------------------------------------------------------------
// Bank generation. bank1 block b: rows 0..15 = Re(psi_j), 16..31 = Im(psi_j),
// tap k -> tau = (Mb - k)/sr (pre-flipped so conv reads the signal forward).
// ---------------------------------------------------------------------------
__global__ void gen_bank1(h16* __restrict__ bank, Cfg1 cfg) {
  int b = blockIdx.z, r = blockIdx.y;
  int k = blockIdx.x * 256 + threadIdx.x;
  int Kb = cfg.Kb[b];
  if (k >= Kb) return;
  int Mb = cfg.Mb[b];
  int j  = 16 * b + (r & 15);
  float f  = 24000.0f * exp2f(-(float)j * (1.0f / 16.0f));
  float Mj = ceilf(3.0f * SRATE / f);
  float d  = (float)(Mb - k);
  float v  = 0.0f;
  if (fabsf(d) <= Mj) {
    float a  = (d / SRATE) * f;
    float g  = PI_M025 * sqrtf(f) * expf(-0.5f * a * a);
    float ph = TWOPI * a;
    v = (r < 16) ? g * cosf(ph) : g * sinf(ph);
  }
  bank[(size_t)cfg.off[b] + (size_t)r * Kb + k] = (h16)v;
}

// bank2 rows are padded: row stride K2s = K2 + 608, tap data origin at q = 32.
__global__ void gen_bank2(h16* __restrict__ bank, Cfg2 cfg) {
  int i = blockIdx.z, r = blockIdx.y;
  int q = blockIdx.x * 256 + threadIdx.x;
  int K2s = cfg.K2s[i];
  if (q >= K2s) return;
  int M = cfg.M2[i];
  int k = q - 32;
  float f = 24000.0f * exp2f(-(float)i);
  float d = (float)(M - k);
  float v = 0.0f;
  if (fabsf(d) <= (float)M) {
    float a  = (d / SRATE) * f;
    float g  = PI_M025 * sqrtf(f) * expf(-0.5f * a * a);
    float ph = TWOPI * a;
    v = (r == 0) ? g * cosf(ph) : g * sinf(ph);
  }
  bank[(size_t)cfg.off[i] + (size_t)r * K2s + q] = (h16)v;
}

// ---------------------------------------------------------------------------
// First-order scattering: scal[bt][j][SPAD+t] = |sum_k psi_j[k] x[t+k-Mb]|.
// Signal window staged into LDS with global_load_async_to_lds_b128.
// ---------------------------------------------------------------------------
#define KC1 512
#define STG1 (KC1 + 104)              // staged halves (incl. 8-half align slack)
__global__ __launch_bounds__(128) void scat1(
    const h16* __restrict__ xh, const h16* __restrict__ bank,
    h16* __restrict__ scal, Cfg1 cfg) {
  __shared__ __align__(16) h16 xs[STG1 + 8];
  const int lane = threadIdx.x & 31, wave = threadIdx.x >> 5;
  const int g = lane >> 4, r = lane & 15;
  const int b = blockIdx.y, bt = blockIdx.z;
  const int Kb = cfg.Kb[b], Mb = cfg.Mb[b];
  const int twg = blockIdx.x * 64;
  const h16* xrow = xh + (size_t)bt * XH_STRIDE + XPAD;
  const h16* A = bank + cfg.off[b];
  const unsigned ldsbase = (unsigned)(uintptr_t)(&xs[0]);

  v8f accre = {}, accim = {};
  for (int kc = 0; kc < Kb; kc += KC1) {
    int base = twg - Mb + kc;
    int base_al = base & ~7;                       // 16B-aligned (halves)
    int sh = base - base_al;
    const h16* gsrc = xrow + base_al;              // always in padded range
    for (int s = threadIdx.x; s < STG1 / 8; s += 128) {
      unsigned ldsa = ldsbase + (unsigned)s * 16u;
      unsigned voff = (unsigned)s * 16u;
      asm volatile("global_load_async_to_lds_b128 %0, %1, %2"
                   :: "v"(ldsa), "v"(voff), "s"(gsrc) : "memory");
    }
    asm volatile("s_wait_asynccnt 0x0" ::: "memory");
    __syncthreads();

    const h16* arow = A + (size_t)r * Kb + kc + 8 * g;
    const h16* irow = A + (size_t)(16 + r) * Kb + kc + 8 * g;
    __builtin_prefetch(arow + KC1, 0, 1);          // global_prefetch_b8
    __builtin_prefetch(irow + KC1, 0, 1);
    int kend = (KC1 < Kb - kc) ? KC1 : (Kb - kc);  // multiple of 32
    for (int kc2 = 0; kc2 < kend; kc2 += 32) {
      union { v16h h; f32x4 q[2]; } ure, uim;
      ure.q[0] = *(const f32x4*)(arow + kc2);
      ure.q[1] = *(const f32x4*)(arow + kc2 + 16);
      uim.q[0] = *(const f32x4*)(irow + kc2);
      uim.q[1] = *(const f32x4*)(irow + kc2 + 16);
      union { v16h h; h16 e[16]; } ub;
      int p0 = sh + wave * 16 + r + kc2 + 8 * g;
#pragma unroll
      for (int e = 0; e < 16; ++e) ub.e[e] = xs[p0 + e + (e >= 8 ? 8 : 0)];
      accre = __builtin_amdgcn_wmma_f32_16x16x32_f16(
          false, ure.h, false, ub.h, (short)0, accre, false, false);
      accim = __builtin_amdgcn_wmma_f32_16x16x32_f16(
          false, uim.h, false, ub.h, (short)0, accim, false, false);
    }
    __syncthreads();
  }
  int t = twg + wave * 16 + r;
#pragma unroll
  for (int v = 0; v < 8; ++v) {
    int w = v + 8 * g;
    float re = accre[v], im = accim[v];
    scal[(size_t)(bt * NW1 + 16 * b + w) * TS + SPAD + t] =
        (h16)sqrtf(re * re + im * im);
  }
}

// ---------------------------------------------------------------------------
// Second-order time conv (one wavelet i per launch).
// Flipped GEMM: A = 16 padded scalogram rows x 32 u-window (vector loads),
// B[u,tt] = psi2[u-tt] staged in LDS by the Tensor Data Mover (2-row 2D tile).
// ---------------------------------------------------------------------------
#define KC2 512
#define W2W (KC2 + 48)                // 560-tap staged wavelet window
__global__ __launch_bounds__(128) void scat2(
    const h16* __restrict__ scal, const h16* __restrict__ bank2,
    h16* __restrict__ y2, int K2, int M2, int K2s, unsigned off2, int nb) {
  __shared__ __align__(16) h16 w2s[2][W2W];
  const int lane = threadIdx.x & 31, wave = threadIdx.x >> 5;
  const int g = lane >> 4, r = lane & 15;
  const int jb = blockIdx.y, bt = blockIdx.z;
  const int t0 = blockIdx.x * 64 + wave * 16;
  const h16* srow_adj = scal + (size_t)(bt * NW1 + 16 * jb + r) * TS
                        + SPAD + t0 - M2;          // pads guarantee in-range
  const h16* wrow = bank2 + off2;                  // padded: data origin q=32
  const unsigned ldsw = (unsigned)(uintptr_t)(&w2s[0][0]);

  v8f accre = {}, accim = {};
  int Ktot = ((K2 + 15 + 31) / 32) * 32;           // u in [0, K2+15)
  for (int uc = 0; uc < Ktot; uc += KC2) {
    if (threadIdx.x < 32) {
      // TDM descriptor: 2D tile, W2W halves x 2 rows, row stride K2s halves.
      unsigned long long ga = (unsigned long long)(uintptr_t)(wrow + 16 + uc);
      i32x4 g0;
      g0[0] = 1;                                   // count=1 valid descriptor
      g0[1] = (int)ldsw;                           // LDS byte address
      g0[2] = (int)(ga & 0xffffffffu);
      g0[3] = (int)(((ga >> 32) & 0x01ffffffu) | 0x80000000u); // type=2
      i32x8 g1;
      g1[0] = (1 << 16);                           // data_size = 2 bytes
      g1[1] = (int)(((unsigned)K2s & 0xffffu) << 16);   // tensor_dim0 lo16
      g1[2] = (int)((((unsigned)K2s >> 16) & 0xffffu) | (2u << 16)); // dim0 hi | tensor_dim1=2
      g1[3] = (int)((unsigned)W2W << 16);          // tile_dim0
      g1[4] = 2;                                   // tile_dim1 = 2
      g1[5] = K2s;                                 // tensor_dim0_stride
      g1[6] = 0; g1[7] = 0;
      i32x4 gz = {};
      asm volatile("tensor_load_to_lds %0, %1, %2, %3"
                   :: "s"(g0), "s"(g1), "s"(gz), "s"(gz) : "memory");
      __builtin_amdgcn_s_wait_tensorcnt(0);
    }
    __syncthreads();

    __builtin_prefetch(srow_adj + uc + KC2, 0, 1);
    int uend = (KC2 < Ktot - uc) ? KC2 : (Ktot - uc);
    for (int u2 = 0; u2 < uend; u2 += 32) {
      const h16* ap = srow_adj + uc + u2 + 8 * g;  // dword-aligned
      union { v16h h; f32x4u q[2]; } ua;
      ua.q[0] = *(const f32x4u*)(ap);
      ua.q[1] = *(const f32x4u*)(ap + 16);
      union { v16h h; h16 e[16]; } ubre, ubim;
      int p0 = u2 + 8 * g - r + 16;
#pragma unroll
      for (int e = 0; e < 16; ++e) {
        ubre.e[e] = w2s[0][p0 + e + (e >= 8 ? 8 : 0)];
        ubim.e[e] = w2s[1][p0 + e + (e >= 8 ? 8 : 0)];
      }
      accre = __builtin_amdgcn_wmma_f32_16x16x32_f16(
          false, ua.h, false, ubre.h, (short)0, accre, false, false);
      accim = __builtin_amdgcn_wmma_f32_16x16x32_f16(
          false, ua.h, false, ubim.h, (short)0, accim, false, false);
    }
    __syncthreads();
  }
  int t = t0 + r;
#pragma unroll
  for (int v = 0; v < 8; ++v) {
    int jj = 16 * jb + v + 8 * g;
    float re = accre[v], im = accim[v];
    float m = sqrtf(re * re + im * im);
    if (jj >= nb) m = 0.0f;                        // reference zero-pads bands
    y2[(size_t)(bt * NW1 + jj) * T_LEN + t] = (h16)m;
  }
}

// ---------------------------------------------------------------------------
// Frequency-axis conv (25 taps, complex) + |.| + 4096x4 boxcar average.
// ---------------------------------------------------------------------------
__global__ __launch_bounds__(256) void freqavg(
    const h16* __restrict__ y2, float* __restrict__ out,
    WF wf, int nb, int chbase) {
  __shared__ float red[256];
  int fb = blockIdx.x % 48, tb = blockIdx.x / 48;
  int c = blockIdx.y, bt = blockIdx.z;
  float s = 0.0f;
  for (int jj4 = 0; jj4 < 4; ++jj4) {
    int jjg = 4 * fb + jj4;
    for (int tt = threadIdx.x; tt < 4096; tt += 256) {
      int t = tb * 4096 + tt;
      float zre = 0.0f, zim = 0.0f;
#pragma unroll
      for (int dd = 0; dd < 25; ++dd) {
        int jsrc = jjg + 12 - dd;
        if (jsrc >= 0 && jsrc < nb) {
          float yv = (float)y2[(size_t)(bt * NW1 + jsrc) * T_LEN + t];
          zre += yv * wf.re[c][dd];
          zim += yv * wf.im[c][dd];
        }
      }
      s += sqrtf(zre * zre + zim * zim);
    }
  }
  red[threadIdx.x] = s;
  __syncthreads();
  for (int o = 128; o > 0; o >>= 1) {
    if (threadIdx.x < (unsigned)o) red[threadIdx.x] += red[threadIdx.x + o];
    __syncthreads();
  }
  if (threadIdx.x == 0) {
    int ch = chbase + c;
    out[(size_t)((bt * 24 + ch) * 48 + fb) * 4 + tb] = red[0] * (1.0f / 16384.0f);
  }
}

// ---------------------------------------------------------------------------
static inline size_t alup(size_t v) { return (v + 255) & ~(size_t)255; }

extern "C" void kernel_launch(void* const* d_in, const int* in_sizes, int n_in,
                              void* d_out, int out_size, void* d_ws, size_t ws_size,
                              hipStream_t stream) {
  (void)in_sizes; (void)n_in; (void)out_size; (void)ws_size;
  const float* x = (const float*)d_in[0];
  float* out = (float*)d_out;
  char* ws = (char*)d_ws;

  // ---- host-side constant geometry ----
  Cfg1 c1; unsigned cur1 = 0;
  for (int b = 0; b < 12; ++b) {
    double f = 24000.0 * pow(2.0, -(16.0 * b + 15.0) / 16.0);
    int M = (int)ceil(3.0 * 48000.0 / f);
    int K = ((2 * M + 1 + 31) / 32) * 32;
    c1.Mb[b] = M; c1.Kb[b] = K; c1.off[b] = cur1;
    cur1 += (unsigned)(32 * K);
  }
  Cfg2 c2; unsigned cur2 = 0;
  for (int i = 0; i < 12; ++i) {
    double f = 24000.0 * pow(2.0, -(double)i);
    int M = (int)ceil(3.0 * 48000.0 / f);
    int K = ((2 * M + 1 + 31) / 32) * 32;
    c2.M2[i] = M; c2.K2[i] = K; c2.K2s[i] = K + 608;
    c2.off[i] = cur2; c2.nb[i] = 16 * i + 1;
    cur2 += (unsigned)(2 * (K + 608));
  }
  WF wf;
  for (int c = 0; c < 2; ++c) {
    double f = 24000.0 * pow(2.0, -(double)c);
    int Mf = (int)ceil(3.0 * 48000.0 / f);
    for (int dd = 0; dd < 25; ++dd) {
      int d = dd - 12;
      double re = 0.0, im = 0.0;
      if (abs(d) <= Mf) {
        double a = ((double)d / 48000.0) * f;
        double gq = 0.7511255444649425 * sqrt(f) * exp(-0.5 * a * a);
        re = gq * cos(2.0 * M_PI * a);
        im = gq * sin(2.0 * M_PI * a);
      }
      wf.re[c][dd] = (float)re; wf.im[c][dd] = (float)im;
    }
  }

  // ---- workspace layout ----
  size_t off_b1 = 0;
  size_t off_b2 = alup(off_b1 + (size_t)cur1 * sizeof(h16));
  size_t off_xh = alup(off_b2 + (size_t)cur2 * sizeof(h16));
  size_t off_sc = alup(off_xh + (size_t)2 * XH_STRIDE * sizeof(h16));
  size_t off_y2 = alup(off_sc + (size_t)2 * NW1 * TS * sizeof(h16));
  h16* bank1 = (h16*)(ws + off_b1);
  h16* bank2 = (h16*)(ws + off_b2);
  h16* xh    = (h16*)(ws + off_xh);
  h16* scal  = (h16*)(ws + off_sc);
  h16* y2    = (h16*)(ws + off_y2);

  // ---- launches (ordered on `stream`) ----
  xcast<<<dim3(2 * 65536 / 256), 256, 0, stream>>>(x, xh);
  int n8 = (int)((size_t)2 * NW1 * TS / 8);
  zscal<<<dim3((n8 + 255) / 256), 256, 0, stream>>>(scal, n8);
  int maxK1 = c1.Kb[11], maxK2s = c2.K2s[11];
  gen_bank1<<<dim3((maxK1 + 255) / 256, 32, 12), 256, 0, stream>>>(bank1, c1);
  gen_bank2<<<dim3((maxK2s + 255) / 256, 2, 12), 256, 0, stream>>>(bank2, c2);
  scat1<<<dim3(T_LEN / 64, 12, 2), 128, 0, stream>>>(xh, bank1, scal, c1);
  for (int i = 0; i < 12; ++i) {
    scat2<<<dim3(T_LEN / 64, i + 1, 2), 128, 0, stream>>>(
        scal, bank2, y2, c2.K2[i], c2.M2[i], c2.K2s[i], c2.off[i], c2.nb[i]);
    freqavg<<<dim3(48 * 4, 2, 2), 256, 0, stream>>>(y2, out, wf, c2.nb[i], 2 * i);
  }
}
